// Net_90993177133316
// MI455X (gfx1250) — compile-verified, matrix operand order checked by reading, not run on hardware
//
#include <hip/hip_runtime.h>
#include <hip/hip_bf16.h>
#include <math.h>
#include <stdint.h>

// ---------------- problem constants (match reference) ----------------
#define NN   10000      // nodes
#define NE   160000     // edges
#define NG   64         // graphs
#define NT   5          // towers
#define FIN  75
#define FOUT 15
#define NL   4
#define CMSG 375        // NT*FIN  (message GEMM N-dim)
#define CPAD 384        // padded to 16
#define KMSG 225        // 3*FIN   (message GEMM K-dim)
#define KPAD 256        // padded to 32
#define KEA  50         // edge-attr features
#define KEAP 64         // padded to 32
#define NENC 75         // encoder out
#define NENCP 80        // padded to 16
#define AVG_DEG_LOG 2.833213344056216f  // log(17)
#define BN_EPS 1e-5f
#define STD_EPS 1e-5f

typedef __attribute__((ext_vector_type(16))) __bf16 v16bf;
typedef __attribute__((ext_vector_type(8)))  float  v8f;
typedef __attribute__((ext_vector_type(4)))  unsigned int u32x4;
typedef __attribute__((ext_vector_type(8)))  unsigned int u32x8;

union Frag { unsigned short u[16]; v16bf v; };

__device__ __forceinline__ unsigned short f2bf(float f) {
  unsigned int u = __float_as_uint(f);
  u += 0x7FFFu + ((u >> 16) & 1u);   // round-to-nearest-even
  return (unsigned short)(u >> 16);
}

// k-index map for 16-bit A/B fragments of V_WMMA_*_16X16X32 (ISA 7.12.2):
// VGPR j holds K pair {2j,2j+1} (j<4) / {16+2(j-4),...} (j>=4); lanes 16-31 add +8.
__device__ __forceinline__ int kmap(int i, int half) {
  int j = i >> 1;
  int base = (j < 4) ? (2 * j) : (16 + 2 * (j - 4));
  return base + half * 8 + (i & 1);
}

// float atomic max/min via signed/unsigned int ordering trick
__device__ __forceinline__ void atomicMaxF(float* a, float v) {
  if (v >= 0.f) atomicMax((int*)a, __float_as_int(v));
  else          atomicMin((unsigned int*)a, __float_as_uint(v));
}
__device__ __forceinline__ void atomicMinF(float* a, float v) {
  if (v >= 0.f) atomicMin((int*)a, __float_as_int(v));
  else          atomicMax((unsigned int*)a, __float_as_uint(v));
}

// ---------------- elementwise / setup kernels ----------------
__global__ void k_fill(float* p, size_t n, float v) {
  size_t i = (size_t)blockIdx.x * blockDim.x + threadIdx.x;
  if (i < n) p[i] = v;
}

// h0 = concat(node_emb[x0], node_emb[x1]) @ pre_lin_W + b   -> [NN,75]
__global__ void k_embed(const int* __restrict__ x, const float* __restrict__ emb,
                        const float* __restrict__ W, const float* __restrict__ B,
                        float* __restrict__ h) {
  int idx = blockIdx.x * blockDim.x + threadIdx.x;
  if (idx >= NN * FIN) return;
  int o = idx % FIN, n = idx / FIN;
  int x0 = x[n * 2], x1 = x[n * 2 + 1];
  float a = B[o];
  for (int c = 0; c < FIN; ++c) a += emb[x0 * FIN + c] * W[c * FIN + o];
  for (int c = 0; c < FIN; ++c) a += emb[x1 * FIN + c] * W[(FIN + c) * FIN + o];
  h[idx] = a;
}

// ea_bf16 [NE,64] = concat(edge_emb[a0], edge_emb[a1]) padded
__global__ void k_build_ea(const int* __restrict__ eattr, const float* __restrict__ eemb,
                           unsigned short* __restrict__ ea) {
  size_t idx = (size_t)blockIdx.x * blockDim.x + threadIdx.x;
  if (idx >= (size_t)NE * KEAP) return;
  int c = (int)(idx & (KEAP - 1));
  size_t e = idx >> 6;
  float v = 0.f;
  if (c < KEA) {
    int a = eattr[e * 2 + (c >= 25)];
    v = eemb[a * 25 + (c % 25)];
  }
  ea[idx] = f2bf(v);
}

__global__ void k_degree(const int* __restrict__ dst, float* __restrict__ cnt) {
  int e = blockIdx.x * blockDim.x + threadIdx.x;
  if (e < NE) atomicAdd(&cnt[dst[e]], 1.f);
}

__global__ void k_node_scalars(const float* __restrict__ cnt, float* __restrict__ amp,
                               float* __restrict__ att) {
  int n = blockIdx.x * blockDim.x + threadIdx.x;
  if (n >= NN) return;
  float logd = logf(fmaxf(cnt[n], 1.f) + 1.f);
  amp[n] = logd / AVG_DEG_LOG;
  att[n] = AVG_DEG_LOG / logd;
}

// pre_W[l] [T,225,75] -> WT_bf16 [384][256] (n-major, zero padded)
__global__ void k_conv_w_pre(const float* __restrict__ preW, unsigned short* __restrict__ WT, int l) {
  int idx = blockIdx.x * blockDim.x + threadIdx.x;
  if (idx >= CPAD * KPAD) return;
  int c = idx & (KPAD - 1), nn = idx >> 8;
  unsigned short v = 0;
  if (nn < CMSG && c < KMSG) {
    int t = nn / FIN, f = nn % FIN;
    v = f2bf(preW[(((size_t)l * NT + t) * KMSG + c) * FIN + f]);
  }
  WT[idx] = v;
}

// edge_enc_W[l] [50,75] -> encWT_bf16 [80][64]
__global__ void k_conv_w_enc(const float* __restrict__ encW, unsigned short* __restrict__ WT, int l) {
  int idx = blockIdx.x * blockDim.x + threadIdx.x;
  if (idx >= NENCP * KEAP) return;
  int c = idx & (KEAP - 1), nn = idx >> 6;
  unsigned short v = 0;
  if (nn < NENC && c < KEA) v = f2bf(encW[((size_t)l * KEA + c) * NENC + nn]);
  WT[idx] = v;
}

// m_in cols 0..149 = bf16(h[dst] , h[src])
__global__ void k_gather_h(const float* __restrict__ h, const int* __restrict__ src,
                           const int* __restrict__ dst, unsigned short* __restrict__ mIn) {
  size_t idx = (size_t)blockIdx.x * blockDim.x + threadIdx.x;
  if (idx >= (size_t)NE * 150) return;
  int c = (int)(idx % 150);
  size_t e = idx / 150;
  int node = (c < FIN) ? dst[e] : src[e];
  int cc = (c < FIN) ? c : c - FIN;
  mIn[e * KPAD + c] = f2bf(h[(size_t)node * FIN + cc]);
}

__global__ void k_agg_init(float* s, float* q, float* mn, float* mx, size_t n) {
  size_t i = (size_t)blockIdx.x * blockDim.x + threadIdx.x;
  if (i >= n) return;
  s[i] = 0.f; q[i] = 0.f; mn[i] = INFINITY; mx[i] = -INFINITY;
}

// ---------------- WMMA: edge encoder  ea[E,64] @ encWT -> m_in[:,150:225] ----------------
__global__ void k_wmma_enc(const unsigned short* __restrict__ ea,
                           const unsigned short* __restrict__ WT,
                           const float* __restrict__ encB,
                           unsigned short* __restrict__ mIn, int l) {
  const int lane = threadIdx.x & 31;
  const int m = lane & 15, half = lane >> 4;
  const int eBase = blockIdx.x * 16;
  const int nBase = blockIdx.y * 16;
  v8f acc = {};
#pragma unroll
  for (int kb = 0; kb < KEAP; kb += 32) {
    Frag a, b;
#pragma unroll
    for (int i = 0; i < 16; ++i) {
      int k = kb + kmap(i, half);
      a.u[i] = ea[(size_t)(eBase + m) * KEAP + k];
      b.u[i] = WT[(size_t)(nBase + m) * KEAP + k];
    }
    acc = __builtin_amdgcn_wmma_f32_16x16x32_bf16(false, a.v, false, b.v,
                                                  (short)0, acc, false, false);
  }
  const int col = nBase + m;
  if (col >= NENC) return;
  const float bias = encB[l * NENC + col];
#pragma unroll
  for (int r = 0; r < 8; ++r) {
    int edge = eBase + r + half * 8;      // D: VGPR r -> row r (lanes0-15) / r+8 (lanes16-31)
    mIn[(size_t)edge * KPAD + 150 + col] = f2bf(acc[r] + bias);
  }
}

// -------- WMMA message GEMM: TDM-staged A tile in LDS, fused scatter reduction --------
// Block = 1 M-tile (16 edges) x 4 waves; each wave covers 6 of 24 N-tiles.
// A tile (16x256 bf16 = 8KB) is DMA'd to LDS once by the Tensor Data Mover, then
// each wave keeps its A fragments in registers across all its N-tiles (24x less
// A traffic than refetching per N-tile). B (WT, 192KB, read-only) stays L2-resident.
__global__ void __launch_bounds__(128)
k_wmma_msgs(const unsigned short* __restrict__ mIn,
            const unsigned short* __restrict__ WT,
            const float* __restrict__ preB,
            const int* __restrict__ dstIdx,
            float* __restrict__ sum, float* __restrict__ sumsq,
            float* __restrict__ mn, float* __restrict__ mx, int l) {
  __shared__ unsigned short sA[16 * KPAD];      // 8 KB staged A tile
  const int lane = threadIdx.x & 31;
  const int m = lane & 15, half = lane >> 4;
  const int eBase = blockIdx.x * 16;

  if (threadIdx.y == 0) {                       // one wave issues the TDM descriptor
    unsigned long long ga = (unsigned long long)(uintptr_t)(mIn + (size_t)eBase * KPAD);
    unsigned int lds = (unsigned int)(uintptr_t)&sA[0];   // flat->LDS: low 32 bits
    u32x4 g0; u32x8 g1;
    g0[0] = 1u;                                           // count=1 (valid user D#)
    g0[1] = lds;                                          // lds_addr (bytes)
    g0[2] = (unsigned int)ga;                             // global_addr[31:0]
    g0[3] = (unsigned int)((ga >> 32) & 0x01FFFFFFu)      // global_addr[56:32]
            | (2u << 30);                                 // type=2 ("image")
    g1[0] = 1u << 16;                                     // data_size=1 -> 2 bytes
    g1[1] = (unsigned int)KPAD << 16;                     // tensor_dim0 = 256
    g1[2] = 16u << 16;                                    // tensor_dim1 = 16
    g1[3] = (unsigned int)KPAD << 16;                     // tile_dim0 = 256
    g1[4] = 16u;                                          // tile_dim1 = 16 (tile_dim2=0)
    g1[5] = (unsigned int)KPAD;                           // tensor_dim0_stride = 256
    g1[6] = 0u; g1[7] = 0u;
    asm volatile("tensor_load_to_lds %0, %1" :: "s"(g0), "s"(g1) : "memory");
    __builtin_amdgcn_s_wait_tensorcnt(0);
  }
  __syncthreads();

  // A fragments for all 8 K-steps, loaded once from LDS (2x ds_load_b128 each)
  Frag afr[8];
#pragma unroll
  for (int kb = 0; kb < 8; ++kb)
#pragma unroll
    for (int i = 0; i < 16; ++i)
      afr[kb].u[i] = sA[m * KPAD + kb * 32 + kmap(i, half)];

  for (int nt = 0; nt < 6; ++nt) {
    const int nBase = ((int)threadIdx.y * 6 + nt) * 16;
    v8f acc = {};
#pragma unroll
    for (int kb = 0; kb < 8; ++kb) {
      Frag b;
#pragma unroll
      for (int i = 0; i < 16; ++i)
        b.u[i] = WT[(size_t)(nBase + m) * KPAD + kb * 32 + kmap(i, half)];
      acc = __builtin_amdgcn_wmma_f32_16x16x32_bf16(false, afr[kb].v, false, b.v,
                                                    (short)0, acc, false, false);
    }
    const int col = nBase + m;
    if (col < CMSG) {
      const int t = col / FIN, f = col % FIN;
      const float bias = preB[((size_t)l * NT + t) * FIN + f];
#pragma unroll
      for (int r = 0; r < 8; ++r) {
        int edge = eBase + r + half * 8;
        float v = acc[r] + bias;
        size_t o = (size_t)dstIdx[edge] * CMSG + col;
        atomicAdd(sum + o, v);
        atomicAdd(sumsq + o, v * v);
        atomicMaxF(mx + o, v);
        atomicMinF(mn + o, v);
      }
    }
  }
}

// sum->mean, sumsq->std, mask min/max where deg==0
__global__ void k_finalize(float* __restrict__ sum, float* __restrict__ sumsq,
                           float* __restrict__ mn, float* __restrict__ mx,
                           const float* __restrict__ cnt) {
  size_t idx = (size_t)blockIdx.x * blockDim.x + threadIdx.x;
  if (idx >= (size_t)NN * CMSG) return;
  size_t n = idx / CMSG;
  float c = cnt[n];
  float d = fmaxf(c, 1.f);
  float mean = sum[idx] / d;
  float var = sumsq[idx] / d - mean * mean;
  sum[idx] = mean;
  sumsq[idx] = sqrtf(fmaxf(var, 0.f) + STD_EPS);
  if (c <= 0.f) { mn[idx] = 0.f; mx[idx] = 0.f; }
}

// post[n,t,fo] = out[n,t,:] @ post_W[l,t,:,fo] with out built on the fly
__global__ void k_post(const float* __restrict__ h, const float* __restrict__ mean,
                       const float* __restrict__ mn, const float* __restrict__ mx,
                       const float* __restrict__ stdv, const float* __restrict__ amp,
                       const float* __restrict__ att, const float* __restrict__ postW,
                       const float* __restrict__ postB, float* __restrict__ post, int l) {
  int idx = blockIdx.x * blockDim.x + threadIdx.x;
  if (idx >= NN * NT * FOUT) return;
  int fo = idx % FOUT, t = (idx / FOUT) % NT, n = idx / (FOUT * NT);
  const float* W = postW + ((size_t)l * NT + t) * 975 * FOUT;
  float a = postB[((size_t)l * NT + t) * FOUT + fo];
  float am = amp[n], at = att[n];
  for (int c = 0; c < FIN; ++c) a += h[(size_t)n * FIN + c] * W[c * FOUT + fo];
  size_t base = (size_t)n * CMSG + t * FIN;
  for (int f = 0; f < FIN; ++f) {
    float vs[4] = { mean[base + f], mn[base + f], mx[base + f], stdv[base + f] };
#pragma unroll
    for (int s = 0; s < 4; ++s) {
      int c1 = 75 + s * 75 + f;           // agg
      float w = W[c1 * FOUT + fo] + am * W[(c1 + 300) * FOUT + fo]
                                  + at * W[(c1 + 600) * FOUT + fo];
      a += vs[s] * w;
    }
  }
  post[(size_t)n * (NT * FOUT) + t * FOUT + fo] = a;
}

// hc = post[N,75] @ lin_W[l] + b
__global__ void k_lin(const float* __restrict__ post, const float* __restrict__ linW,
                      const float* __restrict__ linB, float* __restrict__ hc, int l) {
  int idx = blockIdx.x * blockDim.x + threadIdx.x;
  if (idx >= NN * FIN) return;
  int o = idx % FIN, n = idx / FIN;
  float a = linB[l * FIN + o];
  for (int k = 0; k < FIN; ++k)
    a += post[(size_t)n * FIN + k] * linW[((size_t)l * FIN + k) * FIN + o];
  hc[idx] = a;
}

__global__ void k_bn_stats(const float* __restrict__ hc, float* __restrict__ mu,
                           float* __restrict__ var) {
  __shared__ float ss[256], sq[256];
  int o = blockIdx.x, t = threadIdx.x;
  float s = 0.f, q = 0.f;
  for (int n = t; n < NN; n += 256) { float v = hc[(size_t)n * FIN + o]; s += v; q += v * v; }
  ss[t] = s; sq[t] = q; __syncthreads();
  for (int w = 128; w > 0; w >>= 1) {
    if (t < w) { ss[t] += ss[t + w]; sq[t] += sq[t + w]; }
    __syncthreads();
  }
  if (t == 0) { float m = ss[0] / NN; mu[o] = m; var[o] = sq[0] / NN - m * m; }
}

__global__ void k_bn_apply(const float* __restrict__ hc, const float* __restrict__ mu,
                           const float* __restrict__ var, const float* __restrict__ g,
                           const float* __restrict__ b, float* __restrict__ h, int l) {
  int idx = blockIdx.x * blockDim.x + threadIdx.x;
  if (idx >= NN * FIN) return;
  int o = idx % FIN;
  float v = (hc[idx] - mu[o]) * rsqrtf(var[o] + BN_EPS) * g[l * FIN + o] + b[l * FIN + o];
  h[idx] = fmaxf(v, 0.f);
}

__global__ void k_pool(const float* __restrict__ h, const int* __restrict__ batch,
                       float* __restrict__ pooled) {
  int idx = blockIdx.x * blockDim.x + threadIdx.x;
  if (idx >= NN * FIN) return;
  int c = idx % FIN, n = idx / FIN;
  atomicAdd(&pooled[(size_t)batch[n] * FIN + c], h[idx]);
}

__global__ void k_mlp(const float* __restrict__ in, const float* __restrict__ W,
                      const float* __restrict__ B, float* __restrict__ out,
                      int K, int M, int relu) {
  int idx = blockIdx.x * blockDim.x + threadIdx.x;
  if (idx >= NG * M) return;
  int o = idx % M, g = idx / M;
  float a = B[o];
  for (int k = 0; k < K; ++k) a += in[(size_t)g * K + k] * W[k * M + o];
  out[idx] = relu ? fmaxf(a, 0.f) : a;
}

// ---------------- launcher ----------------
static inline int cdiv(long long a, long long b) { return (int)((a + b - 1) / b); }

extern "C" void kernel_launch(void* const* d_in, const int* in_sizes, int n_in,
                              void* d_out, int out_size, void* d_ws, size_t ws_size,
                              hipStream_t stream) {
  (void)in_sizes; (void)n_in; (void)out_size; (void)ws_size;
  const int*   x        = (const int*)  d_in[0];
  const int*   eindex   = (const int*)  d_in[1];   // [2,E] : row0=src, row1=dst
  const int*   eattr    = (const int*)  d_in[2];
  const int*   batch    = (const int*)  d_in[3];
  const float* node_emb = (const float*)d_in[4];
  const float* edge_emb = (const float*)d_in[5];
  const float* preLinW  = (const float*)d_in[6];
  const float* preLinB  = (const float*)d_in[7];
  const float* encW     = (const float*)d_in[8];
  const float* encB     = (const float*)d_in[9];
  const float* preW     = (const float*)d_in[10];
  const float* preB     = (const float*)d_in[11];
  const float* postW    = (const float*)d_in[12];
  const float* postB    = (const float*)d_in[13];
  const float* linW     = (const float*)d_in[14];
  const float* linB     = (const float*)d_in[15];
  const float* bnG      = (const float*)d_in[16];
  const float* bnB      = (const float*)d_in[17];
  const float* W1 = (const float*)d_in[18]; const float* B1 = (const float*)d_in[19];
  const float* W2 = (const float*)d_in[20]; const float* B2 = (const float*)d_in[21];
  const float* W3 = (const float*)d_in[22]; const float* B3 = (const float*)d_in[23];

  const int* src = eindex;        // edge_index[0]
  const int* dst = eindex + NE;   // edge_index[1]

  // -------- workspace carving (256B aligned) --------
  char* w = (char*)d_ws;
  auto alloc = [&](size_t bytes) { void* p = (void*)w; w += (bytes + 255) & ~(size_t)255; return p; };
  float* h      = (float*)alloc((size_t)NN * FIN * 4);
  float* hc     = (float*)alloc((size_t)NN * FIN * 4);
  float* post   = (float*)alloc((size_t)NN * FIN * 4);
  float* cnt    = (float*)alloc((size_t)NN * 4);
  float* ampA   = (float*)alloc((size_t)NN * 4);
  float* attA   = (float*)alloc((size_t)NN * 4);
  float* mu     = (float*)alloc(FIN * 4);
  float* var    = (float*)alloc(FIN * 4);
  float* pooled = (float*)alloc((size_t)NG * FIN * 4);
  float* g1     = (float*)alloc((size_t)NG * 50 * 4);
  float* g2     = (float*)alloc((size_t)NG * 25 * 4);
  unsigned short* ea    = (unsigned short*)alloc((size_t)NE * KEAP * 2);
  unsigned short* mIn   = (unsigned short*)alloc((size_t)NE * KPAD * 2);
  unsigned short* WT    = (unsigned short*)alloc((size_t)CPAD * KPAD * 2);
  unsigned short* encWT = (unsigned short*)alloc((size_t)NENCP * KEAP * 2);
  float* sumA   = (float*)alloc((size_t)NN * CMSG * 4);
  float* sumsqA = (float*)alloc((size_t)NN * CMSG * 4);
  float* mnA    = (float*)alloc((size_t)NN * CMSG * 4);
  float* mxA    = (float*)alloc((size_t)NN * CMSG * 4);

  const int TB = 256;

  // zero m_in once (covers K-pad cols 225..255 for all layers)
  k_fill<<<cdiv((size_t)NE * KPAD / 2, TB), TB, 0, stream>>>((float*)mIn, (size_t)NE * KPAD / 2, 0.f);
  k_embed<<<cdiv(NN * FIN, TB), TB, 0, stream>>>(x, node_emb, preLinW, preLinB, h);
  k_build_ea<<<cdiv((size_t)NE * KEAP, TB), TB, 0, stream>>>(eattr, edge_emb, ea);
  k_fill<<<cdiv(NN, TB), TB, 0, stream>>>(cnt, NN, 0.f);
  k_degree<<<cdiv(NE, TB), TB, 0, stream>>>(dst, cnt);
  k_node_scalars<<<cdiv(NN, TB), TB, 0, stream>>>(cnt, ampA, attA);

  for (int l = 0; l < NL; ++l) {
    k_conv_w_enc<<<cdiv(NENCP * KEAP, TB), TB, 0, stream>>>(encW, encWT, l);
    k_conv_w_pre<<<cdiv(CPAD * KPAD, TB), TB, 0, stream>>>(preW, WT, l);
    // edge encoder -> m_in[:,150:225] (bf16, +bias)
    k_wmma_enc<<<dim3(NE / 16, NENCP / 16), dim3(32, 1), 0, stream>>>(ea, encWT, encB, mIn, l);
    // h gathers -> m_in[:,0:150]
    k_gather_h<<<cdiv((size_t)NE * 150, TB), TB, 0, stream>>>(h, src, dst, mIn);
    k_agg_init<<<cdiv((size_t)NN * CMSG, TB), TB, 0, stream>>>(sumA, sumsqA, mnA, mxA, (size_t)NN * CMSG);
    // message GEMM (WMMA bf16, TDM-staged A) fused with scatter sum/sumsq/min/max
    k_wmma_msgs<<<dim3(NE / 16), dim3(32, 4), 0, stream>>>(
        mIn, WT, preB, dst, sumA, sumsqA, mnA, mxA, l);
    k_finalize<<<cdiv((size_t)NN * CMSG, TB), TB, 0, stream>>>(sumA, sumsqA, mnA, mxA, cnt);
    k_post<<<cdiv(NN * NT * FOUT, TB), TB, 0, stream>>>(h, sumA, mnA, mxA, sumsqA,
                                                        ampA, attA, postW, postB, post, l);
    k_lin<<<cdiv(NN * FIN, TB), TB, 0, stream>>>(post, linW, linB, hc, l);
    k_bn_stats<<<FIN, 256, 0, stream>>>(hc, mu, var);
    k_bn_apply<<<cdiv(NN * FIN, TB), TB, 0, stream>>>(hc, mu, var, bnG, bnB, h, l);
  }

  k_fill<<<cdiv(NG * FIN, TB), TB, 0, stream>>>(pooled, (size_t)NG * FIN, 0.f);
  k_pool<<<cdiv(NN * FIN, TB), TB, 0, stream>>>(h, batch, pooled);
  k_mlp<<<cdiv(NG * 50, TB), TB, 0, stream>>>(pooled, W1, B1, g1, 75, 50, 1);
  k_mlp<<<cdiv(NG * 25, TB), TB, 0, stream>>>(g1, W2, B2, g2, 50, 25, 1);
  k_mlp<<<cdiv(NG * 1, TB), TB, 0, stream>>>(g2, W3, B3, (float*)d_out, 25, 1, 0);
}
